// GCN_69114613730235
// MI455X (gfx1250) — compile-verified, matrix operand order checked by reading, not run on hardware
//
#include <hip/hip_runtime.h>
#include <hip/hip_bf16.h>

#define N_NODES  100000
#define NFEATS   512
#define NHIDS    256
#define NCLASSES 40
#define MT       (N_NODES / 16)   // 6250 16-row tiles
#define MB       (N_NODES / 32)   // 3125 32-row blocks

typedef __attribute__((ext_vector_type(16))) __bf16 v16bf;
typedef __attribute__((ext_vector_type(8)))  float  v8f;

// K-offset of fragment element e (0..15) for the 16-bit A/B WMMA layout,
// before the per-lane half offset (+8 for lanes 16..31).
// e 0..7 -> K 0..7, e 8..15 -> K 16..23 (contiguous runs of 8).
__device__ __forceinline__ constexpr int kbase(int e) {
  return (((e >> 1) & 3) << 1) + (e & 1) + ((e >> 3) << 4);
}

// ---------------------------------------------------------------------------
// Pack W1[512,256] fp32 -> bf16 B-fragments in exact WMMA B layout.
// Fragment (ntile,kt): lane's 16 elements are 32 contiguous bytes.
// Index: W1p[(ntile*16 + kt)*32 + lane].
// ---------------------------------------------------------------------------
__global__ __launch_bounds__(256) void pack_w1(const float* __restrict__ W1,
                                               v16bf* __restrict__ W1p) {
  const int idx   = blockIdx.x * 256 + threadIdx.x;   // 16*16*32 = 8192
  const int lane  = idx & 31;
  const int kt    = (idx >> 5) & 15;
  const int ntile = idx >> 9;
  const int khalf = (lane >> 4) << 3;
  const int n     = ntile * 16 + (lane & 15);
  v16bf b;
#pragma unroll
  for (int e = 0; e < 16; ++e)
    b[e] = (__bf16)W1[(size_t)(kt * 32 + kbase(e) + khalf) * NHIDS + n];
  W1p[idx] = b;
}

// W2[256,40] -> 3 n-tiles (N padded to 48), 8 k-steps. 768 fragment-lanes.
__global__ __launch_bounds__(256) void pack_w2(const float* __restrict__ W2,
                                               v16bf* __restrict__ W2p) {
  const int idx = blockIdx.x * 256 + threadIdx.x;
  if (idx >= 3 * 8 * 32) return;
  const int lane  = idx & 31;
  const int kt    = (idx >> 5) & 7;
  const int ntile = idx >> 8;
  const int khalf = (lane >> 4) << 3;
  const int n     = ntile * 16 + (lane & 15);
  v16bf b;
#pragma unroll
  for (int e = 0; e < 16; ++e)
    b[e] = (n < NCLASSES)
               ? (__bf16)W2[(size_t)(kt * 32 + kbase(e) + khalf) * NCLASSES + n]
               : (__bf16)0.f;
  W2p[idx] = b;
}

// ---------------------------------------------------------------------------
// Fill an accumulator with a broadcast bias vector (fuses "+ b").
// ---------------------------------------------------------------------------
template <int F>
__global__ __launch_bounds__(256) void init_bias(float* __restrict__ dst,
                                                 const float* __restrict__ bias,
                                                 int total) {
  int i = blockIdx.x * 256 + threadIdx.x;
  if (i < total) dst[i] = bias[i % F];
}

// ---------------------------------------------------------------------------
// GEMM1: H1(bf16)[100000,256] = X[100000,512] * W1[512,256]
// One block (8 waves) per 32-row super-tile (2 m-tiles).  X tile (32x512 fp32
// = 64 KB) staged once in LDS -> X read from HBM exactly once.  Each wave
// owns 2 n-tiles x 2 m-tiles = 4 accumulators; every B fragment feeds two
// WMMAs so its L2 latency hides behind independent work.
// ---------------------------------------------------------------------------
__global__ __launch_bounds__(256) void gemm1_xw1(const float* __restrict__ X,
                                                 const v16bf* __restrict__ W1p,
                                                 __bf16* __restrict__ H1) {
  __shared__ __align__(16) float  xs[32 * NFEATS];   // 64 KB
  __shared__ __align__(16) __bf16 hs[32 * NHIDS];    // 16 KB output staging
  const int t    = threadIdx.x;
  const int mBlk = blockIdx.x;

  // cooperative tile load: 16384 floats = 256 threads x 16 float4
  {
    const float* __restrict__ src = X + (size_t)mBlk * 32 * NFEATS;
#pragma unroll
    for (int i = 0; i < 16; ++i) {
      const int idx = (i * 256 + t) * 4;
      *(float4*)(xs + idx) = *(const float4*)(src + idx);
    }
  }
  __syncthreads();

  const int wv    = t >> 5;            // wave 0..7 -> n-tiles {2wv, 2wv+1}
  const int lane  = t & 31;
  const int khalf = (lane >> 4) << 3;
  const int lrow  = lane & 15;
  const float* __restrict__ xr0 = xs + lrow * NFEATS;          // m-subtile 0
  const float* __restrict__ xr1 = xs + (16 + lrow) * NFEATS;   // m-subtile 1
  // per-wave fragment base pointers: per-kt offset folds into imm offset
  const v16bf* __restrict__ bp0 = W1p + (wv * 2 + 0) * 16 * 32 + lane;
  const v16bf* __restrict__ bp1 = W1p + (wv * 2 + 1) * 16 * 32 + lane;

  v8f acc[2][2] = {{v8f{}, v8f{}}, {v8f{}, v8f{}}};  // [msub][ntile]
#pragma unroll 4
  for (int kt = 0; kt < 16; ++kt) {
    v16bf a0, a1;
    {   // A fragments: two contiguous 8-float runs per lane -> 4x ds_load_b128
      const float4 f0 = *(const float4*)(xr0 + kt * 32 + khalf);
      const float4 f1 = *(const float4*)(xr0 + kt * 32 + khalf + 4);
      const float4 f2 = *(const float4*)(xr0 + kt * 32 + 16 + khalf);
      const float4 f3 = *(const float4*)(xr0 + kt * 32 + 16 + khalf + 4);
      a0[0]  = (__bf16)f0.x; a0[1]  = (__bf16)f0.y; a0[2]  = (__bf16)f0.z; a0[3]  = (__bf16)f0.w;
      a0[4]  = (__bf16)f1.x; a0[5]  = (__bf16)f1.y; a0[6]  = (__bf16)f1.z; a0[7]  = (__bf16)f1.w;
      a0[8]  = (__bf16)f2.x; a0[9]  = (__bf16)f2.y; a0[10] = (__bf16)f2.z; a0[11] = (__bf16)f2.w;
      a0[12] = (__bf16)f3.x; a0[13] = (__bf16)f3.y; a0[14] = (__bf16)f3.z; a0[15] = (__bf16)f3.w;
    }
    {
      const float4 f0 = *(const float4*)(xr1 + kt * 32 + khalf);
      const float4 f1 = *(const float4*)(xr1 + kt * 32 + khalf + 4);
      const float4 f2 = *(const float4*)(xr1 + kt * 32 + 16 + khalf);
      const float4 f3 = *(const float4*)(xr1 + kt * 32 + 16 + khalf + 4);
      a1[0]  = (__bf16)f0.x; a1[1]  = (__bf16)f0.y; a1[2]  = (__bf16)f0.z; a1[3]  = (__bf16)f0.w;
      a1[4]  = (__bf16)f1.x; a1[5]  = (__bf16)f1.y; a1[6]  = (__bf16)f1.z; a1[7]  = (__bf16)f1.w;
      a1[8]  = (__bf16)f2.x; a1[9]  = (__bf16)f2.y; a1[10] = (__bf16)f2.z; a1[11] = (__bf16)f2.w;
      a1[12] = (__bf16)f3.x; a1[13] = (__bf16)f3.y; a1[14] = (__bf16)f3.z; a1[15] = (__bf16)f3.w;
    }
    const v16bf b0 = bp0[kt * 32];   // const imm offset, 2x global b128
    const v16bf b1 = bp1[kt * 32];
    acc[0][0] = __builtin_amdgcn_wmma_f32_16x16x32_bf16(
        false, a0, false, b0, (short)0, acc[0][0], false, false);
    acc[1][0] = __builtin_amdgcn_wmma_f32_16x16x32_bf16(
        false, a1, false, b0, (short)0, acc[1][0], false, false);
    acc[0][1] = __builtin_amdgcn_wmma_f32_16x16x32_bf16(
        false, a0, false, b1, (short)0, acc[0][1], false, false);
    acc[1][1] = __builtin_amdgcn_wmma_f32_16x16x32_bf16(
        false, a1, false, b1, (short)0, acc[1][1], false, false);
  }

  // stage D tiles as bf16 in LDS, then copy out coalesced (b128 stores)
#pragma unroll
  for (int ms = 0; ms < 2; ++ms) {
#pragma unroll
    for (int s = 0; s < 2; ++s) {
      const int n = (wv * 2 + s) * 16 + lrow;
#pragma unroll
      for (int d = 0; d < 8; ++d)
        hs[(ms * 16 + khalf + d) * NHIDS + n] = (__bf16)acc[ms][s][d];
    }
  }
  __syncthreads();
  {
    uint4*       dst = (uint4*)(H1 + (size_t)mBlk * 32 * NHIDS);
    const uint4* src = (const uint4*)hs;
#pragma unroll
    for (int j = 0; j < 4; ++j)       // 16 KB tile: 256 thr x 64 B
      dst[4 * t + j] = src[4 * t + j];
  }
}

// ---------------------------------------------------------------------------
// SpMM1: acc2[r,:] += val * H1[c,:]   (one wave per edge, 256 feats)
// H1 bf16 (51 MB) is L2-resident; accumulate with native f32 atomics.
// ---------------------------------------------------------------------------
__global__ __launch_bounds__(256) void spmm_h1(const int* __restrict__ erow,
                                               const int* __restrict__ ecol,
                                               const float* __restrict__ ev,
                                               const __bf16* __restrict__ H1,
                                               float* __restrict__ acc2,
                                               int nE) {
  const int e    = (blockIdx.x * blockDim.x + threadIdx.x) >> 5;
  const int lane = threadIdx.x & 31;
  if (e >= nE) return;
  const int   r = erow[e];
  const int   c = ecol[e];
  const float v = ev[e];
  const __bf16* __restrict__ src = H1 + (size_t)c * NHIDS;
  float* __restrict__ dst = acc2 + (size_t)r * NHIDS;
  union { uint4 u; __bf16 h[8]; } U;            // one b128 = 8 bf16 per lane
  U.u = *(const uint4*)(src + lane * 8);
#pragma unroll
  for (int i = 0; i < 8; ++i) {
    const float xv = (float)U.h[i];
    __hip_atomic_fetch_add(dst + lane * 8 + i, v * xv,
                           __ATOMIC_RELAXED, __HIP_MEMORY_SCOPE_AGENT);
  }
}

// ---------------------------------------------------------------------------
// GEMM2: H3[100000,40] = relu(acc2) * W2   (one wave per m-tile, 3 n-tiles)
// A rows are read exactly once -> direct global float4 loads, fused ReLU.
// ---------------------------------------------------------------------------
__global__ __launch_bounds__(256) void gemm2_hw2(const float* __restrict__ acc2,
                                                 const v16bf* __restrict__ W2p,
                                                 float* __restrict__ H3) {
  const int wid  = (blockIdx.x * blockDim.x + threadIdx.x) >> 5;
  const int lane = threadIdx.x & 31;
  if (wid >= MT) return;                  // wave-uniform: EXEC stays all-1s
  const int khalf = (lane >> 4) << 3;
  const int lrow  = lane & 15;
  const float* __restrict__ hrow = acc2 + (size_t)(wid * 16 + lrow) * NHIDS;
  const v16bf* __restrict__ bp = W2p + lane;   // per-kt offset -> imm offset

  v8f acc[3] = {v8f{}, v8f{}, v8f{}};
#pragma unroll
  for (int kt = 0; kt < 8; ++kt) {
    float4 f[4];
    f[0] = *(const float4*)(hrow + kt * 32 + khalf);
    f[1] = *(const float4*)(hrow + kt * 32 + khalf + 4);
    f[2] = *(const float4*)(hrow + kt * 32 + 16 + khalf);
    f[3] = *(const float4*)(hrow + kt * 32 + 16 + khalf + 4);
    v16bf a;
#pragma unroll
    for (int q = 0; q < 4; ++q) {
      a[q * 4 + 0] = (__bf16)(f[q].x > 0.f ? f[q].x : 0.f);   // fused ReLU
      a[q * 4 + 1] = (__bf16)(f[q].y > 0.f ? f[q].y : 0.f);
      a[q * 4 + 2] = (__bf16)(f[q].z > 0.f ? f[q].z : 0.f);
      a[q * 4 + 3] = (__bf16)(f[q].w > 0.f ? f[q].w : 0.f);
    }
#pragma unroll
    for (int s = 0; s < 3; ++s) {
      const v16bf b = bp[(s * 8 + kt) * 32];
      acc[s] = __builtin_amdgcn_wmma_f32_16x16x32_bf16(
          false, a, false, b, (short)0, acc[s], false, false);
    }
  }
  const int mrow = wid * 16 + khalf;
#pragma unroll
  for (int s = 0; s < 3; ++s) {
    const int n = s * 16 + lrow;
    if (n < NCLASSES) {
#pragma unroll
      for (int d = 0; d < 8; ++d)
        H3[(size_t)(mrow + d) * NCLASSES + n] = acc[s][d];
    }
  }
}

// ---------------------------------------------------------------------------
// SpMM2: out[r,:] += val * H3[c,:]  (one wave per edge, 40 feats, L2-resident)
// ---------------------------------------------------------------------------
__global__ __launch_bounds__(256) void spmm_out(const int* __restrict__ erow,
                                                const int* __restrict__ ecol,
                                                const float* __restrict__ ev,
                                                const float* __restrict__ H3,
                                                float* __restrict__ out,
                                                int nE) {
  const int e    = (blockIdx.x * blockDim.x + threadIdx.x) >> 5;
  const int lane = threadIdx.x & 31;
  if (e >= nE) return;
  const int   r = erow[e];
  const int   c = ecol[e];
  const float v = ev[e];
  const float* __restrict__ src = H3 + (size_t)c * NCLASSES;
  float* __restrict__ dst = out + (size_t)r * NCLASSES;
  __hip_atomic_fetch_add(dst + lane, v * src[lane],
                         __ATOMIC_RELAXED, __HIP_MEMORY_SCOPE_AGENT);
  if (lane < NCLASSES - 32)
    __hip_atomic_fetch_add(dst + 32 + lane, v * src[32 + lane],
                           __ATOMIC_RELAXED, __HIP_MEMORY_SCOPE_AGENT);
}

// ---------------------------------------------------------------------------
extern "C" void kernel_launch(void* const* d_in, const int* in_sizes, int n_in,
                              void* d_out, int out_size, void* d_ws, size_t ws_size,
                              hipStream_t stream) {
  const float* x   = (const float*)d_in[0];
  const float* W1  = (const float*)d_in[1];
  const float* b1  = (const float*)d_in[2];
  const float* W2  = (const float*)d_in[3];
  const float* b2  = (const float*)d_in[4];
  const int*   er  = (const int*)d_in[5];
  const int*   ec  = (const int*)d_in[6];
  const float* ev  = (const float*)d_in[7];
  float*       out = (float*)d_out;
  const int    nE  = in_sizes[5];

  char* ws = (char*)d_ws;
  __bf16* H1   = (__bf16*)ws;                                   // 51.2 MB
  float*  acc2 = (float*)(ws + (size_t)N_NODES * NHIDS * 2);    // 102.4 MB
  float*  H3   = (float*)(ws + (size_t)N_NODES * NHIDS * 6);    // 16 MB
  v16bf*  W1p  = (v16bf*)(ws + (size_t)N_NODES * NHIDS * 6 +
                          (size_t)N_NODES * NCLASSES * 4);      // 256 KB
  v16bf*  W2p  = W1p + 16 * 16 * 32;                            // 24 KB

  // one-time weight repack into WMMA B-fragment layout (bf16)
  pack_w1<<<32, 256, 0, stream>>>(W1, W1p);
  pack_w2<<<3, 256, 0, stream>>>(W2, W2p);

  // init accumulators with biases (fuses +b1 / +b2)
  init_bias<NHIDS><<<(N_NODES * NHIDS) / 256, 256, 0, stream>>>(
      acc2, b1, N_NODES * NHIDS);
  init_bias<NCLASSES><<<(N_NODES * NCLASSES + 255) / 256, 256, 0, stream>>>(
      out, b2, N_NODES * NCLASSES);

  // layer 1 dense: H1 = bf16(X @ W1)
  gemm1_xw1<<<MB, 256, 0, stream>>>(x, W1p, H1);
  // acc2 += A-gather of H1
  spmm_h1<<<(nE + 7) / 8, 256, 0, stream>>>(er, ec, ev, H1, acc2, nE);
  // layer 2 dense: H3 = relu(acc2) @ W2
  gemm2_hw2<<<(MT * 32 + 255) / 256, 256, 0, stream>>>(acc2, W2p, H3);
  // out(=b2) += A-gather of H3
  spmm_out<<<(nE + 7) / 8, 256, 0, stream>>>(er, ec, ev, H3, out, nE);
}
